// Dilated_UNet_config1_21646635171980
// MI455X (gfx1250) — matrix operand order repaired and checked by reading, not compiled, not run on hardware
//
#include <hip/hip_runtime.h>
#include <hip/hip_bf16.h>

// Dilated U-Net forward for MI455X (gfx1250, wave32, WMMA).
//
// Roofline: ~100 GFLOP of conv work vs ~1 GB of f16 activation traffic at
// 23.3 TB/s -> bandwidth-bound. Design:
//  * intermediate activations in HBM as f16 (halves traffic, direct d16
//    loads into WMMA B fragments)
//  * all matmul work on v_wmma_f32_16x16x32_f16 (f32 accumulate)
//  * weights pre-packed ONCE per launch into f16 in the exact ISA A-fragment
//    lane layout -> each lane's A fragment is one contiguous 32B global load
//    (L2-resident, ~3.9 MB total) at an immediate offset; no LDS staging, no
//    per-chunk barriers in the conv hot loop
//  * im2col K-order = (kh,kw) outer, cin inner; CIN%32==0 so each K-chunk of
//    32 shares one (kh,kw): one bounds check + base address per chunk, then
//    16 constant-offset d16 loads per lane
//  * wave tile = 32 couts x 32 pixels (2x2 register blocking, 4 WMMAs/chunk);
//    block = 8 waves = 32 couts x 256 pixels; all geometry is template-const
//    so pixel decode is shifts and offsets fold into immediates
//  * conv + bias + ReLU + BN-statistics fused; training-mode BN applied in a
//    second elementwise pass after the global channel reduction
//  * bias slice staged to LDS via CDNA5 async global->LDS copy when the
//    toolchain exposes the builtins (ASYNCcnt-tracked), else plain load
//
// Workspace requirement: ~450 MB of d_ws.

typedef __attribute__((ext_vector_type(16))) _Float16 v16h;
typedef __attribute__((ext_vector_type(8)))  float    v8f;

#define NB 8
#define HH 256
#define WW 256

#if defined(__has_builtin)
#if __has_builtin(__builtin_amdgcn_global_load_async_to_lds_b32) && \
    __has_builtin(__builtin_amdgcn_s_wait_asynccnt)
#define USE_ASYNC_LDS 1
#endif
#endif

// ---------------------------------------------------------------------------
// Weight pre-pack: f32 OIHW -> f16 fragment-ready blocks.
// Layout: [mt][kh*KSZ+kw][cb][512] where each 512-half block is [lane][16]
// in the ISA 16-bit A-matrix lane layout:
//   VGPR v<4  -> K = hi*8 + v*2 + odd ; VGPR v>=4 -> K = 16 + hi*8 + (v-4)*2 + odd
// ---------------------------------------------------------------------------
__global__ void pack_weights(const float* __restrict__ w, _Float16* __restrict__ o,
                             int CIN, int KSZ, int total) {
    int d = blockIdx.x * 256 + threadIdx.x;
    if (d >= total) return;
    int KKl = KSZ * KSZ;
    int CB  = CIN >> 5;
    int h = d & 15;
    int l = (d >> 4) & 31;
    int rest = d >> 9;
    int cb = rest % CB; rest /= CB;
    int p  = rest % KKl;
    int mt = rest / KKl;
    int v = h >> 1, odd = h & 1, hi = l >> 4;
    int kk = (v < 4) ? (hi * 8 + v * 2 + odd)
                     : (16 + hi * 8 + (v - 4) * 2 + odd);
    int m   = mt * 16 + (l & 15);
    int cin = cb * 32 + kk;
    o[d] = (_Float16)w[(m * CIN + cin) * KKl + p];
}

// ---------------------------------------------------------------------------
// Implicit-GEMM conv via WMMA. grid.x = B*HOUT*WOUT/256, grid.y = COUT/32.
// in: f16 activations, wpk: pre-packed f16 weights, out: f16 relu(conv+bias),
// stats: per-channel sum / sum^2 (f32) for the following BN pass.
// ---------------------------------------------------------------------------
template <int CIN, int COUT, int KSZ, int PAD, int DIL,
          int HIN, int WIN, int HOUT, int WOUT>
__global__ __launch_bounds__(256, 2)
void conv_wmma(const _Float16* __restrict__ in, const _Float16* __restrict__ wpk,
               const float* __restrict__ bias, _Float16* __restrict__ out,
               float* __restrict__ stats) {
    constexpr int KK     = KSZ * KSZ;
    constexpr int CB     = CIN / 32;
    constexpr int NCHUNK = KK * CB;
    constexpr int HWI    = HIN * WIN;
    constexpr int HWO    = HOUT * WOUT;

    __shared__ float ssum[32];
    __shared__ float ssq[32];
    __shared__ float sbias[32];

    const int tid  = threadIdx.x;
    const int lane = tid & 31;
    const int wave = tid >> 5;
    const int khi  = lane >> 4;            // K-half: lanes 0-15 / 16-31
    const int m0   = blockIdx.y * 32;

    // Stage the block's 32 bias values into LDS (async copy if available).
#ifdef USE_ASYNC_LDS
    if (tid < 32)
        __builtin_amdgcn_global_load_async_to_lds_b32(
            (__attribute__((address_space(1))) int*)(bias + m0 + tid),
            (__attribute__((address_space(3))) int*)&sbias[tid], 0, 0);
#else
    if (tid < 32) sbias[tid] = bias[m0 + tid];
#endif
    if (tid < 32) { ssum[tid] = 0.f; ssq[tid] = 0.f; }

    // Two pixel tiles per wave; HWO/WOUT powers of two -> shifts only.
    int bimg[2], oy[2], ox[2];
#pragma unroll
    for (int u = 0; u < 2; ++u) {
        int nl  = blockIdx.x * 256 + wave * 32 + u * 16 + (lane & 15);
        bimg[u] = nl / HWO;
        int rem = nl - bimg[u] * HWO;
        oy[u]   = rem / WOUT;
        ox[u]   = rem - oy[u] * WOUT;
    }

    // Per-lane A fragment bases: consecutive chunks are +512 halves apart.
    const _Float16* a0p = wpk + ((size_t)(blockIdx.y * 2 + 0) * NCHUNK) * 512 + lane * 16;
    const _Float16* a1p = wpk + ((size_t)(blockIdx.y * 2 + 1) * NCHUNK) * 512 + lane * 16;

    v8f acc[2][2] = {};
    int chunk = 0;
    for (int kh = 0; kh < KSZ; ++kh)
      for (int kw = 0; kw < KSZ; ++kw)
        for (int cb = 0; cb < CIN; cb += 32, ++chunk) {
            v16h a0 = *(const v16h*)(a0p + (size_t)chunk * 512);
            v16h a1 = *(const v16h*)(a1p + (size_t)chunk * 512);

#pragma unroll
            for (int u = 0; u < 2; ++u) {
                // One bounds check + base address per chunk; 16 d16 loads at
                // compile-time constant offsets ((h + (h >= 8 ? 8 : 0)) * HWI * 2 bytes < 8 MB).
                int iy = oy[u] + kh * DIL - PAD;
                int ix = ox[u] + kw * DIL - PAD;
                v16h bf = {};
                if (iy >= 0 && iy < HIN && ix >= 0 && ix < WIN) {
                    const _Float16* bp =
                        in + ((long)(bimg[u] * CIN + cb + khi * 8) * HIN + iy) * WIN + ix;
                    __builtin_prefetch(bp + (long)32 * HWI, 0, 1);  // next cin block
#pragma unroll
                    for (int h = 0; h < 16; ++h) bf[h] = bp[(long)(h + (h >= 8 ? 8 : 0)) * HWI];
                }
                acc[0][u] = __builtin_amdgcn_wmma_f32_16x16x32_f16(
                    false, a0, false, bf, (short)0, acc[0][u], false, false);
                acc[1][u] = __builtin_amdgcn_wmma_f32_16x16x32_f16(
                    false, a1, false, bf, (short)0, acc[1][u], false, false);
            }
        }

#ifdef USE_ASYNC_LDS
    __builtin_amdgcn_s_wait_asynccnt(0);
#endif
    __syncthreads();

    // Epilogue: bias + ReLU, f16 store, BN statistics.
    // C/D layout: lanes 0-15 -> M=r, lanes 16-31 -> M=8+r, N = lane&15.
#pragma unroll
    for (int t = 0; t < 2; ++t) {
#pragma unroll
        for (int r = 0; r < 8; ++r) {
            int   ml = t * 16 + r + khi * 8;
            float bs = sbias[ml];
#pragma unroll
            for (int u = 0; u < 2; ++u) {
                float v = acc[t][u][r] + bs;
                v = v > 0.f ? v : 0.f;
                out[((long)(bimg[u] * COUT + m0 + ml) * HOUT + oy[u]) * WOUT + ox[u]] =
                    (_Float16)v;
                atomicAdd(&ssum[ml], v);
                atomicAdd(&ssq[ml], v * v);
            }
        }
    }
    __syncthreads();
    if (tid < 32) {
        atomicAdd(&stats[m0 + tid],        ssum[tid]);
        atomicAdd(&stats[COUT + m0 + tid], ssq[tid]);
    }
}

// ---------------------------------------------------------------------------
// Training-mode BN apply (in place, f16): y = (y-mean)*rsqrt(var+eps)*g + be
// ---------------------------------------------------------------------------
__global__ void bn_apply(_Float16* __restrict__ y, const float* __restrict__ stats,
                         const float* __restrict__ g, const float* __restrict__ be,
                         int C, int HW, float invcnt, long total) {
    long i = (long)blockIdx.x * 256 + threadIdx.x;
    if (i >= total) return;
    int c = (int)((i / HW) % C);
    float mean = stats[c] * invcnt;
    float var  = stats[C + c] * invcnt - mean * mean;
    float s    = rsqrtf(var + 1e-5f);
    float yv   = (float)y[i];
    y[i] = (_Float16)((yv - mean) * s * g[c] + be[c]);
}

// ---------------------------------------------------------------------------
// Dynamic per-sample conv: x[B,1,256,256] (*) w[B,32,4,4], pad=1 -> [B,32,255,255]
// Fused ReLU + BN stats. grid: x = pixel tiles, y = b*32+c.
// ---------------------------------------------------------------------------
__global__ void dyn_conv(const float* __restrict__ x, const float* __restrict__ w,
                         _Float16* __restrict__ out, float* __restrict__ stats,
                         int H, int W, int Ho, int Wo) {
    int bc  = blockIdx.y;
    int b   = bc >> 5, c = bc & 31;
    int pix = blockIdx.x * 256 + threadIdx.x;
    __shared__ float s2[2];
    __shared__ float wk[16];
    if (threadIdx.x < 2)  s2[threadIdx.x] = 0.f;
    if (threadIdx.x < 16) wk[threadIdx.x] = w[(bc << 4) + threadIdx.x];
    __syncthreads();
    if (pix < Ho * Wo) {
        int oy = pix / Wo, ox = pix - oy * Wo;
        float s = 0.f;
#pragma unroll
        for (int kh = 0; kh < 4; ++kh)
#pragma unroll
            for (int kw = 0; kw < 4; ++kw) {
                int iy = oy + kh - 1, ix = ox + kw - 1;
                if (iy >= 0 && iy < H && ix >= 0 && ix < W)
                    s += x[((long)b * H + iy) * W + ix] * wk[kh * 4 + kw];
            }
        float v = s > 0.f ? s : 0.f;
        out[((long)bc * Ho + oy) * Wo + ox] = (_Float16)v;
        atomicAdd(&s2[0], v);
        atomicAdd(&s2[1], v * v);
    }
    __syncthreads();
    if (threadIdx.x == 0) {
        atomicAdd(&stats[c],      s2[0]);
        atomicAdd(&stats[32 + c], s2[1]);
    }
}

__global__ void maxpool2(const _Float16* __restrict__ in, _Float16* __restrict__ out,
                         int Hi, int Wi, long total) {
    long i = (long)blockIdx.x * 256 + threadIdx.x;
    if (i >= total) return;
    int Ho = Hi >> 1, Wo = Wi >> 1;
    long hw = (long)Ho * Wo;
    long bc = i / hw;
    long r  = i - bc * hw;
    int oy  = (int)(r / Wo), ox = (int)(r - (long)oy * Wo);
    const _Float16* p = in + (bc * Hi + oy * 2) * Wi + ox * 2;
    float a  = fmaxf((float)p[0], (float)p[1]);
    float b2 = fmaxf((float)p[Wi], (float)p[Wi + 1]);
    out[i] = (_Float16)fmaxf(a, b2);
}

__global__ void up2_concat(const _Float16* __restrict__ up, const _Float16* __restrict__ skip,
                           _Float16* __restrict__ out, int Cup, int Cskip,
                           int Ho, int Wo, long total) {
    long i = (long)blockIdx.x * 256 + threadIdx.x;
    if (i >= total) return;
    int C = Cup + Cskip;
    long hw = (long)Ho * Wo;
    long bcl = i / hw;
    long r   = i - bcl * hw;
    int b = (int)(bcl / C);
    int c = (int)(bcl - (long)b * C);
    int y = (int)(r / Wo), x = (int)(r - (long)y * Wo);
    _Float16 v;
    if (c < Cup)
        v = up[((long)(b * Cup + c) * (Ho >> 1) + (y >> 1)) * (Wo >> 1) + (x >> 1)];
    else
        v = skip[((long)(b * Cskip + (c - Cup)) * Ho + y) * Wo + x];
    out[i] = v;
}

__global__ void conv1x1_head(const _Float16* __restrict__ in, const float* __restrict__ w,
                             const float* __restrict__ bias, float* __restrict__ out,
                             int Cin, int HW, long total) {
    long i = (long)blockIdx.x * 256 + threadIdx.x;   // over B*HW
    if (i >= total) return;
    int b = (int)(i / HW);
    int p = (int)(i - (long)b * HW);
#pragma unroll
    for (int o = 0; o < 2; ++o) {
        float s = bias[o];
        for (int c = 0; c < Cin; ++c)
            s += (float)in[((long)b * Cin + c) * HW + p] * w[o * Cin + c];
        out[((long)b * 2 + o) * HW + p] = s;
    }
}

// ---------------------------------------------------------------------------
// Host orchestration
// ---------------------------------------------------------------------------
template <int CIN, int COUT, int KSZ, int PAD, int DIL,
          int HIN, int WIN, int HOUT, int WOUT>
static void launch_conv(const _Float16* in, const _Float16* wpk, const float* bias,
                        _Float16* out, float* stats, hipStream_t s) {
    int NTOT = NB * HOUT * WOUT;
    dim3 grid((unsigned)(NTOT / 256), COUT / 32);
    conv_wmma<CIN, COUT, KSZ, PAD, DIL, HIN, WIN, HOUT, WOUT>
        <<<grid, 256, 0, s>>>(in, wpk, bias, out, stats);
}

extern "C" void kernel_launch(void* const* d_in, const int* in_sizes, int n_in,
                              void* d_out, int out_size, void* d_ws, size_t ws_size,
                              hipStream_t stream) {
    (void)in_sizes; (void)n_in; (void)out_size; (void)ws_size;
    const float* x = (const float*)d_in[0];
    const float* w = (const float*)d_in[1];
    auto P = [&](int i) { return (const float*)d_in[i]; };

    // ---- workspace layout (bump allocator, 256B aligned) ----
    char*  ws  = (char*)d_ws;
    size_t off = 0;
    auto allocf = [&](size_t n) {
        float* p = (float*)(ws + off);
        off = (off + n * sizeof(float) + 255) & ~(size_t)255;
        return p;
    };
    auto alloch = [&](size_t n) {
        _Float16* p = (_Float16*)(ws + off);
        off = (off + n * sizeof(_Float16) + 255) & ~(size_t)255;
        return p;
    };

    float* stats = allocf(14 * 512);                 // 14 BN slots x (sum,sumsq)
    _Float16* t0   = alloch((size_t)NB * 32 * 255 * 255);
    _Float16* enc0 = alloch((size_t)NB * 32 * 256 * 256);
    _Float16* p0   = alloch((size_t)NB * 32 * 128 * 128);
    _Float16* t1   = alloch((size_t)NB * 64 * 128 * 128);
    _Float16* enc1 = alloch((size_t)NB * 64 * 128 * 128);
    _Float16* p1   = alloch((size_t)NB * 64 * 64 * 64);
    _Float16* t2   = alloch((size_t)NB * 128 * 64 * 64);
    _Float16* enc2 = alloch((size_t)NB * 128 * 64 * 64);
    _Float16* p2   = alloch((size_t)NB * 128 * 32 * 32);
    _Float16* m1   = alloch((size_t)NB * 256 * 32 * 32);
    _Float16* m2   = alloch((size_t)NB * 256 * 32 * 32);
    _Float16* cat1 = alloch((size_t)NB * 384 * 64 * 64);
    _Float16* u1a  = alloch((size_t)NB * 128 * 64 * 64);
    _Float16* u1b  = alloch((size_t)NB * 128 * 64 * 64);
    _Float16* cat2 = alloch((size_t)NB * 192 * 128 * 128);
    _Float16* u2a  = alloch((size_t)NB * 64 * 128 * 128);
    _Float16* u2b  = alloch((size_t)NB * 64 * 128 * 128);
    _Float16* cat3 = alloch((size_t)NB * 96 * 256 * 256);
    _Float16* u3a  = alloch((size_t)NB * 32 * 256 * 256);
    _Float16* u3b  = alloch((size_t)NB * 32 * 256 * 256);

    auto ST = [&](int i) { return stats + (size_t)i * 512; };
    (void)hipMemsetAsync(stats, 0, 14 * 512 * sizeof(float), stream);

    // ---- pre-pack all conv weights (f32 OIHW -> f16 fragment blocks) ----
    auto pack = [&](const float* wsrc, int COUT, int CIN, int KSZ) {
        int total = COUT * CIN * KSZ * KSZ;
        _Float16* dst = alloch((size_t)total);
        pack_weights<<<dim3((unsigned)((total + 255) / 256)), 256, 0, stream>>>(
            wsrc, dst, CIN, KSZ, total);
        return dst;
    };
    _Float16* W_d1  = pack(P(4),  32,  32,  4);
    _Float16* W_d2a = pack(P(8),  64,  32,  3);
    _Float16* W_d2b = pack(P(12), 64,  64,  3);
    _Float16* W_d3a = pack(P(16), 128, 64,  3);
    _Float16* W_d3b = pack(P(20), 128, 128, 3);
    _Float16* W_mda = pack(P(24), 256, 128, 3);
    _Float16* W_mdb = pack(P(28), 256, 256, 3);
    _Float16* W_u1a = pack(P(32), 128, 384, 3);
    _Float16* W_u1b = pack(P(36), 128, 128, 3);
    _Float16* W_u2a = pack(P(40), 64,  192, 3);
    _Float16* W_u2b = pack(P(44), 64,  64,  3);
    _Float16* W_u3a = pack(P(48), 32,  96,  3);
    _Float16* W_u3b = pack(P(52), 32,  32,  3);

    auto bn = [&](_Float16* y, float* st, const float* g, const float* be, int C, int HW) {
        long total = (long)NB * C * HW;
        bn_apply<<<dim3((unsigned)((total + 255) / 256)), 256, 0, stream>>>(
            y, st, g, be, C, HW, 1.f / ((float)NB * HW), total);
    };
    auto mp = [&](const _Float16* in, _Float16* out, int C, int Hi) {
        long total = (long)NB * C * (Hi / 2) * (Hi / 2);
        maxpool2<<<dim3((unsigned)((total + 255) / 256)), 256, 0, stream>>>(
            in, out, Hi, Hi, total);
    };
    auto cat = [&](const _Float16* up, const _Float16* skip, _Float16* out,
                   int Cup, int Cskip, int Ho) {
        long total = (long)NB * (Cup + Cskip) * Ho * Ho;
        up2_concat<<<dim3((unsigned)((total + 255) / 256)), 256, 0, stream>>>(
            up, skip, out, Cup, Cskip, Ho, Ho, total);
    };

    // ---- encoder ----
    dyn_conv<<<dim3((255 * 255 + 255) / 256, NB * 32), 256, 0, stream>>>(
        x, w, t0, ST(0), HH, WW, 255, 255);
    bn(t0, ST(0), P(2), P(3), 32, 255 * 255);
    launch_conv<32, 32, 4, 5, 3, 255, 255, 256, 256>(t0, W_d1, P(5), enc0, ST(1), stream);
    bn(enc0, ST(1), P(6), P(7), 32, 256 * 256);
    mp(enc0, p0, 32, 256);

    launch_conv<32, 64, 3, 1, 1, 128, 128, 128, 128>(p0, W_d2a, P(9), t1, ST(2), stream);
    bn(t1, ST(2), P(10), P(11), 64, 128 * 128);
    launch_conv<64, 64, 3, 1, 1, 128, 128, 128, 128>(t1, W_d2b, P(13), enc1, ST(3), stream);
    bn(enc1, ST(3), P(14), P(15), 64, 128 * 128);
    mp(enc1, p1, 64, 128);

    launch_conv<64, 128, 3, 1, 1, 64, 64, 64, 64>(p1, W_d3a, P(17), t2, ST(4), stream);
    bn(t2, ST(4), P(18), P(19), 128, 64 * 64);
    launch_conv<128, 128, 3, 1, 1, 64, 64, 64, 64>(t2, W_d3b, P(21), enc2, ST(5), stream);
    bn(enc2, ST(5), P(22), P(23), 128, 64 * 64);
    mp(enc2, p2, 128, 64);

    // ---- bottleneck ----
    launch_conv<128, 256, 3, 1, 1, 32, 32, 32, 32>(p2, W_mda, P(25), m1, ST(6), stream);
    bn(m1, ST(6), P(26), P(27), 256, 32 * 32);
    launch_conv<256, 256, 3, 1, 1, 32, 32, 32, 32>(m1, W_mdb, P(29), m2, ST(7), stream);
    bn(m2, ST(7), P(30), P(31), 256, 32 * 32);

    // ---- decoder ----
    cat(m2, enc2, cat1, 256, 128, 64);
    launch_conv<384, 128, 3, 1, 1, 64, 64, 64, 64>(cat1, W_u1a, P(33), u1a, ST(8), stream);
    bn(u1a, ST(8), P(34), P(35), 128, 64 * 64);
    launch_conv<128, 128, 3, 1, 1, 64, 64, 64, 64>(u1a, W_u1b, P(37), u1b, ST(9), stream);
    bn(u1b, ST(9), P(38), P(39), 128, 64 * 64);

    cat(u1b, enc1, cat2, 128, 64, 128);
    launch_conv<192, 64, 3, 1, 1, 128, 128, 128, 128>(cat2, W_u2a, P(41), u2a, ST(10), stream);
    bn(u2a, ST(10), P(42), P(43), 64, 128 * 128);
    launch_conv<64, 64, 3, 1, 1, 128, 128, 128, 128>(u2a, W_u2b, P(45), u2b, ST(11), stream);
    bn(u2b, ST(11), P(46), P(47), 64, 128 * 128);

    cat(u2b, enc0, cat3, 64, 32, 256);
    launch_conv<96, 32, 3, 1, 1, 256, 256, 256, 256>(cat3, W_u3a, P(49), u3a, ST(12), stream);
    bn(u3a, ST(12), P(50), P(51), 32, 256 * 256);
    launch_conv<32, 32, 3, 1, 1, 256, 256, 256, 256>(u3a, W_u3b, P(53), u3b, ST(13), stream);
    bn(u3b, ST(13), P(54), P(55), 32, 256 * 256);

    // ---- head ----
    long htot = (long)NB * 256 * 256;
    conv1x1_head<<<dim3((unsigned)((htot + 255) / 256)), 256, 0, stream>>>(
        u3b, P(56), P(57), (float*)d_out, 32, 256 * 256, htot);
}
